// SMRCAEncoderBlock_4647154614906
// MI455X (gfx1250) — matrix operand order repaired and checked by reading, not compiled, not run on hardware
//
#include <hip/hip_runtime.h>
#include <hip/hip_bf16.h>

typedef __attribute__((ext_vector_type(16))) __bf16 bf16x16;
typedef __attribute__((ext_vector_type(8)))  __bf16 bf16x8;
typedef __attribute__((ext_vector_type(8)))  float  f32x8;
typedef __attribute__((ext_vector_type(4)))  int    v4i;

#define TMB 128   // block tile M
#define TN  128   // block tile N
#define TK  32    // K step
#define APAD 40   // pad K-rows 32 -> 40 elements (80B) to spread LDS banks
#define BPAD 40

#if defined(__HIP_DEVICE_COMPILE__) && __has_builtin(__builtin_amdgcn_global_load_async_to_lds_b128) && __has_builtin(__builtin_amdgcn_s_wait_asynccnt)
#define USE_ASYNC_LDS 1
#define GLOBAL_AS __attribute__((address_space(1)))
#define LDS_AS    __attribute__((address_space(3)))
#define WAIT_ASYNC() __builtin_amdgcn_s_wait_asynccnt(0)
#else
#define USE_ASYNC_LDS 0
#define WAIT_ASYNC()
#endif

__device__ __forceinline__ __bf16 f2bf(float f) {
    unsigned u = __builtin_bit_cast(unsigned, f);
    u += 0x7FFFu + ((u >> 16) & 1u);           // round-to-nearest-even
    unsigned short h = (unsigned short)(u >> 16);
    return __builtin_bit_cast(__bf16, h);
}

__device__ __forceinline__ float gelu_f(float x) {
    const float k0 = 0.7978845608028654f, k1 = 0.044715f;
    return 0.5f * x * (1.0f + tanhf(k0 * (x + k1 * x * x * x)));
}

// ---------------------------------------------------------------------------
// f32 -> bf16 elementwise convert (weight prep)
// ---------------------------------------------------------------------------
__global__ void conv_bf16_kernel(const float* __restrict__ src, __bf16* __restrict__ dst, int n) {
    int i = blockIdx.x * 256 + threadIdx.x;
    if (i < n) dst[i] = f2bf(src[i]);
}

// ---------------------------------------------------------------------------
// GroupNorm over [B,C,NP]; one block per (batch, group); bf16 output
// ---------------------------------------------------------------------------
__global__ __launch_bounds__(256)
void groupnorm_bf16_kernel(const float* __restrict__ x, const float* __restrict__ gamma,
                           const float* __restrict__ beta, __bf16* __restrict__ out,
                           int C, int groups, int NP) {
    const int cg = C / groups;
    const int g  = blockIdx.x % groups;
    const int b  = blockIdx.x / groups;
    const long base = ((long)b * C + (long)g * cg) * NP;
    const int  n    = cg * NP;

    float s = 0.f, ss = 0.f;
    for (int i = threadIdx.x; i < n; i += 256) {
        float v = x[base + i];
        s += v; ss += v * v;
    }
    __shared__ float rs[256], rq[256];
    rs[threadIdx.x] = s; rq[threadIdx.x] = ss;
    __syncthreads();
    for (int off = 128; off > 0; off >>= 1) {
        if ((int)threadIdx.x < off) {
            rs[threadIdx.x] += rs[threadIdx.x + off];
            rq[threadIdx.x] += rq[threadIdx.x + off];
        }
        __syncthreads();
    }
    const float mean = rs[0] / (float)n;
    const float var  = rq[0] / (float)n - mean * mean;
    const float inv  = rsqrtf(var + 1e-5f);

    for (int i = threadIdx.x; i < n; i += 256) {
        int c = g * cg + i / NP;
        float v = (x[base + i] - mean) * inv;
        out[base + i] = f2bf(v * gamma[c] + beta[c]);
    }
}

// ---------------------------------------------------------------------------
// bf16 WMMA GEMM: out[b, M, NP] = W[M,K] @ X[b, K, NP] + bias (+epilogue)
// MODE 0: f32 out; MODE 1: f32 out + residual; MODE 2: bf16 out + GELU
// Block tile 128x128, 8 waves (4M x 2N), wave tile 32x64 (8 WMMA / K-step),
// K step 32, double-buffered LDS; A tile via async global->LDS (ASYNCcnt),
// B tile transposed to [n][k] in LDS for 2x ds_load_b128 fragments.
// ---------------------------------------------------------------------------
template <int MODE>
__global__ __launch_bounds__(256)
void gemm_bf16_kernel(const __bf16* __restrict__ W, const __bf16* __restrict__ X,
                      const float* __restrict__ bias0, const float* __restrict__ bias1, int bsplit,
                      const float* __restrict__ residual,
                      float* __restrict__ outF, __bf16* __restrict__ outB,
                      int M, int K, int NP) {
    __shared__ alignas(16) __bf16 lA[2][TMB][APAD];  // [m][k], row-major
    __shared__ alignas(16) __bf16 lB[2][TN][BPAD];   // [n][k], k contiguous per column

    const int b     = blockIdx.z;
    const int mBase = blockIdx.y * TMB;
    const int nBase = blockIdx.x * TN;
    const int tid   = threadIdx.x;
    const int wid   = tid >> 5;
    const int lane  = tid & 31;
    const int ln    = lane & 15;
    const int half  = lane >> 4;
    const int wm    = wid & 3;     // 4 M-spans of 32 (2 tiles each)
    const int wn    = wid >> 2;    // 2 N-spans of 64 (4 tiles each)

    const __bf16* Xb = X + (long)b * K * NP;

    f32x8 acc[2][4] = {};

    auto fillA = [&](int buf, int ks) {
        int r  = tid >> 1;           // 0..127
        int kq = (tid & 1) * 16;     // 0 or 16
        const __bf16* src = W + (long)(mBase + r) * K + ks + kq;
#if USE_ASYNC_LDS
        __builtin_amdgcn_global_load_async_to_lds_b128(
            (GLOBAL_AS v4i*)src, (LDS_AS v4i*)&lA[buf][r][kq], 0, 0);
        __builtin_amdgcn_global_load_async_to_lds_b128(
            (GLOBAL_AS v4i*)(src + 8), (LDS_AS v4i*)&lA[buf][r][kq + 8], 0, 0);
#else
        *(bf16x8*)&lA[buf][r][kq]     = *(const bf16x8*)src;
        *(bf16x8*)&lA[buf][r][kq + 8] = *(const bf16x8*)(src + 8);
#endif
    };
    auto fillB = [&](int buf, int ks) {
        int k  = tid >> 3;           // 0..31
        int n0 = (tid & 7) * 16;     // 0..112
        const __bf16* src = Xb + (long)(ks + k) * NP + nBase + n0;
        bf16x8 v0 = *(const bf16x8*)(src);
        bf16x8 v1 = *(const bf16x8*)(src + 8);
#pragma unroll
        for (int e = 0; e < 8; ++e) lB[buf][n0 + e][k] = v0[e];
#pragma unroll
        for (int e = 0; e < 8; ++e) lB[buf][n0 + 8 + e][k] = v1[e];
    };
    auto compute = [&](int buf) {
        union ufrag { bf16x16 f; bf16x8 h[2]; };
        ufrag A[2], Bf[4];
        // Load ALL fragments first so one dscnt wait covers everything and the
        // 8 WMMAs issue back-to-back (disjoint accumulators -> no hazards).
#pragma unroll
        for (int mt = 0; mt < 2; ++mt) {
            const __bf16* ar = &lA[buf][wm * 32 + mt * 16 + ln][0];
            A[mt].h[0] = *(const bf16x8*)(ar + half * 8);        // K 0..7  / 8..15
            A[mt].h[1] = *(const bf16x8*)(ar + 16 + half * 8);   // K 16..23 / 24..31
        }
#pragma unroll
        for (int nt = 0; nt < 4; ++nt) {
            const __bf16* br = &lB[buf][wn * 64 + nt * 16 + ln][0];
            Bf[nt].h[0] = *(const bf16x8*)(br + half * 8);
            Bf[nt].h[1] = *(const bf16x8*)(br + 16 + half * 8);
        }
#pragma unroll
        for (int mt = 0; mt < 2; ++mt)
#pragma unroll
            for (int nt = 0; nt < 4; ++nt)
                acc[mt][nt] = __builtin_amdgcn_wmma_f32_16x16x32_bf16(
                    false, A[mt].f, false, Bf[nt].f, (short)0, acc[mt][nt], false, false);
    };

    const int S = K / TK;
    fillA(0, 0);
    fillB(0, 0);
    for (int s = 0; s < S; ++s) {
        WAIT_ASYNC();          // own async A-tile loads done before signaling
        __syncthreads();       // all waves' fills for buffer s visible
        if (s + 2 < S) {       // global_prefetch_b8 the K+2 tiles, keep near
            __builtin_prefetch(W + (long)(mBase + (tid >> 1)) * K + (s + 2) * TK, 0, 3);
            __builtin_prefetch(Xb + (long)((s + 2) * TK + (tid >> 3)) * NP + nBase, 0, 3);
        }
        if (s + 1 < S) {
            fillA((s + 1) & 1, (s + 1) * TK);
            fillB((s + 1) & 1, (s + 1) * TK);
        }
        compute(s & 1);
    }

    // Epilogue: C layout — VGPR r: lanes 0-15 -> M=r, lanes 16-31 -> M=r+8; N=lane&15
#pragma unroll
    for (int mt = 0; mt < 2; ++mt) {
#pragma unroll
        for (int nt = 0; nt < 4; ++nt) {
#pragma unroll
            for (int r = 0; r < 8; ++r) {
                int m = mBase + wm * 32 + mt * 16 + r + 8 * half;
                int n = nBase + wn * 64 + nt * 16 + ln;
                long oidx = ((long)b * M + m) * NP + n;
                float v = acc[mt][nt][r] + ((m < bsplit) ? bias0[m] : bias1[m - bsplit]);
                if constexpr (MODE == 1) v += residual[oidx];
                if constexpr (MODE == 2) {
                    outB[oidx] = f2bf(gelu_f(v));
                } else {
                    outF[oidx] = v;
                }
            }
        }
    }
}

// ---------------------------------------------------------------------------
// Sparse multi-range attention: 3x3 window, dilation 4*(r+1), h_per=2, d=32
// One thread per (batch, head, pixel). q:[b,256,NP], kv:[b,512,NP] (k|v).
// ---------------------------------------------------------------------------
__global__ __launch_bounds__(256)
void attn_kernel(const float* __restrict__ q, const float* __restrict__ kv,
                 __bf16* __restrict__ out) {
    const int p = blockIdx.x * 256 + threadIdx.x;  // 0..9215
    const int h = blockIdx.y;                      // 0..7
    const int b = blockIdx.z;                      // 0..1
    const int y = p / 96, x = p % 96;
    const int r = h >> 1;
    const int dil = 4 * (r + 1);

    const long qbase = ((long)b * 256 + h * 32) * 9216 + p;
    const long kbase = ((long)b * 512 + h * 32) * 9216;
    const long vbase = ((long)b * 512 + 256 + h * 32) * 9216;
    const float scale = 0.17677669529663687f;  // 1/sqrt(32)

    float qv[32];
#pragma unroll
    for (int c = 0; c < 32; ++c) qv[c] = q[qbase + (long)c * 9216];

    float s[9];
    float smax = -1e30f;
#pragma unroll
    for (int t = 0; t < 9; ++t) {
        int yy = y + (t / 3 - 1) * dil;
        int xx = x + (t % 3 - 1) * dil;
        float a = 0.f;
        if ((unsigned)yy < 96u && (unsigned)xx < 96u) {
            long pp = yy * 96 + xx;
#pragma unroll
            for (int c = 0; c < 32; ++c) a += qv[c] * kv[kbase + (long)c * 9216 + pp];
        }
        s[t] = a * scale;   // OOB taps score 0 (zero-padded window), still in softmax
        smax = fmaxf(smax, s[t]);
    }
    float denom = 0.f;
#pragma unroll
    for (int t = 0; t < 9; ++t) { s[t] = __expf(s[t] - smax); denom += s[t]; }
    const float inv = 1.f / denom;

    float accv[32];
#pragma unroll
    for (int c = 0; c < 32; ++c) accv[c] = 0.f;
#pragma unroll
    for (int t = 0; t < 9; ++t) {
        int yy = y + (t / 3 - 1) * dil;
        int xx = x + (t % 3 - 1) * dil;
        if ((unsigned)yy < 96u && (unsigned)xx < 96u) {
            long pp = yy * 96 + xx;
            float w = s[t] * inv;
#pragma unroll
            for (int c = 0; c < 32; ++c) accv[c] += w * kv[vbase + (long)c * 9216 + pp];
        }
    }
#pragma unroll
    for (int c = 0; c < 32; ++c)
        out[((long)b * 256 + h * 32 + c) * 9216 + p] = f2bf(accv[c]);
}

// ---------------------------------------------------------------------------
// Launch: GN -> q/kv GEMMs -> attention -> o GEMM(+res) -> GN2 -> mlp1(GELU)
//         -> mlp2(+res -> d_out). Workspace regions aliased over time.
// ---------------------------------------------------------------------------
extern "C" void kernel_launch(void* const* d_in, const int* in_sizes, int n_in,
                              void* d_out, int out_size, void* d_ws, size_t ws_size,
                              hipStream_t stream) {
    const float* x_q    = (const float*)d_in[0];
    const float* x_kv   = (const float*)d_in[1];
    const float* gnq_w  = (const float*)d_in[2];
    const float* gnq_b  = (const float*)d_in[3];
    const float* gnkv_w = (const float*)d_in[4];
    const float* gnkv_b = (const float*)d_in[5];
    const float* gn2_w  = (const float*)d_in[6];
    const float* gn2_b  = (const float*)d_in[7];
    const float* wq = (const float*)d_in[8];   const float* bq = (const float*)d_in[9];
    const float* wk = (const float*)d_in[10];  const float* bk = (const float*)d_in[11];
    const float* wv = (const float*)d_in[12];  const float* bv = (const float*)d_in[13];
    const float* wo = (const float*)d_in[14];  const float* bo = (const float*)d_in[15];
    const float* w1 = (const float*)d_in[16];  const float* b1 = (const float*)d_in[17];
    const float* w2 = (const float*)d_in[18];  const float* b2 = (const float*)d_in[19];
    float* out = (float*)d_out;

    char* ws = (char*)d_ws;
    // region A (9.44MB): qn|kvn, later attn (bf16)
    __bf16* qn    = (__bf16*)(ws + 0);
    __bf16* kvn   = (__bf16*)(ws + 4718592);
    __bf16* attnb = (__bf16*)(ws + 0);
    // region B (18.87MB): qbuf (f32), later xbuf (f32, first half)
    float* qbuf = (float*)(ws + 9437184);
    float* xbuf = (float*)(ws + 9437184);
    // region C (37.75MB): kvbuf (f32), later hn (bf16) + h1 (bf16)
    float*  kvbuf = (float*)(ws + 28311552);
    __bf16* hn    = (__bf16*)(ws + 28311552);
    __bf16* h1b   = (__bf16*)(ws + 28311552 + 4718592);
    // weights (bf16)
    __bf16* wqb  = (__bf16*)(ws + 66060288);
    __bf16* wkvb = (__bf16*)(ws + 66060288 + 65536);
    __bf16* wob  = (__bf16*)(ws + 66060288 + 196608);
    __bf16* w1b  = (__bf16*)(ws + 66060288 + 262144);
    __bf16* w2b  = (__bf16*)(ws + 66060288 + 393216);

    dim3 blk(256);

    // weight conversion f32 -> bf16 (wk|wv fused into one [512,128] matrix)
    conv_bf16_kernel<<<128, blk, 0, stream>>>(wq, wqb, 32768);
    conv_bf16_kernel<<<128, blk, 0, stream>>>(wk, wkvb, 32768);
    conv_bf16_kernel<<<128, blk, 0, stream>>>(wv, wkvb + 32768, 32768);
    conv_bf16_kernel<<<128, blk, 0, stream>>>(wo, wob, 32768);
    conv_bf16_kernel<<<256, blk, 0, stream>>>(w1, w1b, 65536);
    conv_bf16_kernel<<<256, blk, 0, stream>>>(w2, w2b, 65536);

    // GroupNorm on q and kv inputs
    groupnorm_bf16_kernel<<<16, blk, 0, stream>>>(x_q,  gnq_w,  gnq_b,  qn,  128, 8, 9216);
    groupnorm_bf16_kernel<<<16, blk, 0, stream>>>(x_kv, gnkv_w, gnkv_b, kvn, 128, 8, 9216);

    // q = wq @ qn + bq   (M=256,K=128)   -> qbuf
    gemm_bf16_kernel<0><<<dim3(72, 2, 2), blk, 0, stream>>>(
        wqb, qn, bq, bq, 256, nullptr, qbuf, nullptr, 256, 128, 9216);
    // [k;v] = [wk;wv] @ kvn + [bk;bv]  (M=512,K=128) -> kvbuf
    gemm_bf16_kernel<0><<<dim3(72, 4, 2), blk, 0, stream>>>(
        wkvb, kvn, bk, bv, 256, nullptr, kvbuf, nullptr, 512, 128, 9216);

    // sparse multi-range attention -> attnb (bf16, reuses qn/kvn region)
    attn_kernel<<<dim3(36, 8, 2), blk, 0, stream>>>(qbuf, kvbuf, attnb);

    // x = x_q + wo @ attn + bo   (M=128,K=256) -> xbuf (reuses qbuf region)
    gemm_bf16_kernel<1><<<dim3(72, 1, 2), blk, 0, stream>>>(
        wob, attnb, bo, bo, 128, x_q, xbuf, nullptr, 128, 256, 9216);

    // GN2(x) -> hn (bf16, reuses kvbuf region)
    groupnorm_bf16_kernel<<<16, blk, 0, stream>>>(xbuf, gn2_w, gn2_b, hn, 128, 8, 9216);

    // h1 = gelu(w1 @ hn + b1)  (M=512,K=128) -> h1b (bf16)
    gemm_bf16_kernel<2><<<dim3(72, 4, 2), blk, 0, stream>>>(
        w1b, hn, b1, b1, 512, nullptr, nullptr, h1b, 512, 128, 9216);

    // out = x + w2 @ h1 + b2   (M=128,K=512) -> d_out
    gemm_bf16_kernel<1><<<dim3(72, 1, 2), blk, 0, stream>>>(
        w2b, h1b, b2, b2, 128, xbuf, out, nullptr, 128, 512, 9216);
}
